// decoder_80607946211782
// MI455X (gfx1250) — matrix operand order repaired and checked by reading, not compiled, not run on hardware
//
#include <hip/hip_runtime.h>
#include <hip/hip_bf16.h>

#define BTOT 1024
#define TM1  127
#define ENCD 256
#define DECD 256

typedef __attribute__((ext_vector_type(16))) __bf16 v16bf;
typedef __attribute__((ext_vector_type(8)))  float  v8f;

// ---------------- bf16 helpers ----------------
// Native conversion: lowers to hardware v_cvt (packed where possible).
__device__ __forceinline__ unsigned short f2bf(float f) {
  __bf16 b = (__bf16)f;
  return __builtin_bit_cast(unsigned short, b);
}
__device__ __forceinline__ float bf2f(unsigned short s) {
  return __uint_as_float(((unsigned int)s) << 16);
}
__device__ __forceinline__ float2 bf2x2(unsigned int w) {
  float2 r;
  r.x = __uint_as_float(w << 16);
  r.y = __uint_as_float(w & 0xFFFF0000u);
  return r;
}
__device__ __forceinline__ unsigned int pack2bf(float lo, float hi) {
  return (unsigned int)f2bf(lo) | ((unsigned int)f2bf(hi) << 16);
}
__device__ __forceinline__ v8f zero8() {
  v8f v;
#pragma unroll
  for (int i = 0; i < 8; ++i) v[i] = 0.0f;
  return v;
}
__device__ __forceinline__ v16bf frag_from_u8(uint4 a, uint4 b) {
  union { unsigned int u[8]; v16bf v; } r;
  r.u[0] = a.x; r.u[1] = a.y; r.u[2] = a.z; r.u[3] = a.w;
  r.u[4] = b.x; r.u[5] = b.y; r.u[6] = b.z; r.u[7] = b.w;
  return r.v;
}
// A-matrix fragment (16x32 bf16). Lane holds row M=lane&15;
// elems 0..15 -> K = half*16 + j (contiguous, matching B).
__device__ __forceinline__ v16bf afrag_bf16(const unsigned short* rowk, int half) {
  uint4 lo = *(const uint4*)(rowk + half * 16);
  uint4 hi = *(const uint4*)(rowk + half * 16 + 8);
  return frag_from_u8(lo, hi);
}
__device__ __forceinline__ v16bf afrag_f32(const float* rowk, int half) {
  union { unsigned int u[8]; v16bf v; } r;
#pragma unroll
  for (int j = 0; j < 4; ++j)
    r.u[j] = pack2bf(rowk[half * 16 + 2 * j], rowk[half * 16 + 2 * j + 1]);
#pragma unroll
  for (int j = 0; j < 4; ++j)
    r.u[4 + j] = pack2bf(rowk[half * 16 + 8 + 2 * j], rowk[half * 16 + 8 + 2 * j + 1]);
  return r.v;
}
// B-matrix fragment (32x16 bf16). Lane holds column N=lane&15; 16 consecutive
// K values starting at kbase + (lane>=16 ? 16 : 0). Source is W[n][k] row-major.
__device__ __forceinline__ v16bf bfrag_bf16(const unsigned short* colk) {
  uint4 lo = *(const uint4*)(colk);
  uint4 hi = *(const uint4*)(colk + 8);
  return frag_from_u8(lo, hi);
}

// ---------------- fast transcendentals (hardware v_exp/v_rcp) -------------
__device__ __forceinline__ float fast_tanh(float x) {
  float e = __builtin_amdgcn_exp2f(x * 2.88539008178f);   // exp(2x)
  return 1.0f - 2.0f * __builtin_amdgcn_rcpf(1.0f + e);
}
__device__ __forceinline__ float fast_sigmoid(float x) {
  return __builtin_amdgcn_rcpf(1.0f + __builtin_amdgcn_exp2f(-1.44269504089f * x));
}
__device__ __forceinline__ float fast_exp(float x) {
  return __builtin_amdgcn_exp2f(x * 1.44269504089f);
}

// ---------------- kernel 1: fp32 -> bf16 conversion (vectorized x8) -------
// n must be a multiple of 8 (true for all uses here).
__global__ void f32_to_bf16_kernel(const float* __restrict__ src,
                                   unsigned short* __restrict__ dst, int n8) {
  int i = blockIdx.x * blockDim.x + threadIdx.x;
  int stride = gridDim.x * blockDim.x;
  const float4* s4 = (const float4*)src;
  uint4* d4 = (uint4*)dst;
  for (; i < n8; i += stride) {
    float4 a = s4[2 * i];
    float4 b = s4[2 * i + 1];
    uint4 o;
    o.x = pack2bf(a.x, a.y);
    o.y = pack2bf(a.z, a.w);
    o.z = pack2bf(b.x, b.y);
    o.w = pack2bf(b.z, b.w);
    d4[i] = o;
  }
}

// ---------------- kernel 2: Ue = enc @ W_Ud^T + b_Ud (bf16 in+out, WMMA) ---
// rows = B*TM1 = 130048 = 8128 tiles of 16. 128 threads = 4 waves; wave w
// owns columns [w*64, w*64+64) = 4 N-tiles, K=256 in 8 steps of 32.
// Pure bf16 loads -> wmma: no conversion VALU in the inner loop.
__global__ __launch_bounds__(128)
void ue_gemm_kernel(const unsigned short* __restrict__ enc_bf,
                    const unsigned short* __restrict__ WUd_bf,
                    const float* __restrict__ b_Ud,
                    unsigned short* __restrict__ Ue_bf) {
  const int rowTile = blockIdx.x;
  const int wid  = threadIdx.x >> 5;
  const int lane = threadIdx.x & 31;
  const int col  = lane & 15;
  const int half = lane >> 4;

  const unsigned short* arow = enc_bf + (size_t)(rowTile * 16 + col) * ENCD;
  v8f acc[4];
#pragma unroll
  for (int nt = 0; nt < 4; ++nt) acc[nt] = zero8();

  for (int kb = 0; kb < ENCD; kb += 32) {
    v16bf a = afrag_bf16(arow + kb, half);
#pragma unroll
    for (int nt = 0; nt < 4; ++nt) {
      int n = wid * 64 + nt * 16 + col;
      v16bf b = bfrag_bf16(WUd_bf + (size_t)n * ENCD + kb + half * 16);
      acc[nt] = __builtin_amdgcn_wmma_f32_16x16x32_bf16(
          false, a, false, b, (short)0, acc[nt], false, false);
    }
  }
#pragma unroll
  for (int nt = 0; nt < 4; ++nt) {
    int n = wid * 64 + nt * 16 + col;
    float bias = b_Ud[n];
#pragma unroll
    for (int r = 0; r < 8; ++r) {
      int m = r + half * 8;
      Ue_bf[(size_t)(rowTile * 16 + m) * ENCD + n] = f2bf(acc[nt][r] + bias);
    }
  }
}

// ---------------- kernel 3: persistent fused decoder ----------------------
// 64 blocks x 256 threads (8 waves). Block b owns batch rows [16b, 16b+16)
// and runs the full 127-step recurrence; its Ue/enc bf16 slices (2 MB) stay
// L2-resident across steps (64 x 2 MB = 128 MB < 192 MB L2).
__global__ __launch_bounds__(256)
void decoder_kernel(const float* __restrict__ y_history,
                    const unsigned short* __restrict__ Ue_bf,
                    const unsigned short* __restrict__ enc_bf,
                    const unsigned short* __restrict__ Wdhs_bf,  // [ENCD][2*DECD]
                    const unsigned short* __restrict__ Whh_bf,   // [4*DECD][DECD]
                    const float* __restrict__ W_vd, const float* __restrict__ b_vd,
                    const float* __restrict__ b_dhs,
                    const float* __restrict__ W_w,  const float* __restrict__ b_w,
                    const float* __restrict__ W_ih, const float* __restrict__ b_ih,
                    const float* __restrict__ b_hh,
                    const float* __restrict__ W_fc, const float* __restrict__ b_fc,
                    float* __restrict__ out) {
  __shared__ __align__(16) unsigned short h_bf[16][DECD];   // 8 KB  (h, bf16)
  __shared__ __align__(16) float          c_s[16][DECD];    // 16 KB (c, fp32)
  __shared__ __align__(16) unsigned short Wd_bf[16][ENCD];  // 8 KB
  __shared__ __align__(16) float          sc[16][128];      // 8 KB  scores/beta
  __shared__ __align__(16) float          ctx[16][ENCD];    // 16 KB
  __shared__             float            ytil[16];
  __shared__ __align__(16) float          vv[ENCD];         // 1 KB  (v_d)

  const int tid  = threadIdx.x;
  const int lane = tid & 31;
  const int wid  = tid >> 5;
  const int col  = lane & 15;
  const int half = lane >> 4;
  const int r0   = blockIdx.x * 16;

  for (int i = tid; i < 16 * DECD; i += 256) {
    h_bf[i >> 8][i & 255] = 0;
    c_s[i >> 8][i & 255]  = 0.0f;
  }
  if (tid < ENCD) vv[tid] = W_vd[tid];
  __syncthreads();

  const float bvd = b_vd[0];

  for (int s = 0; s < TM1; ++s) {
    // ---- Phase A: Wd = [h,c] @ W_dhs^T + b_dhs  (WMMA, wave owns 2 N-tiles)
    {
      v8f acc[2];
      acc[0] = zero8(); acc[1] = zero8();
      for (int kb = 0; kb < DECD; kb += 32) {                 // h part, k in [0,256)
        v16bf a = afrag_bf16(&h_bf[col][0] + kb, half);
#pragma unroll
        for (int nt = 0; nt < 2; ++nt) {
          int n = wid * 32 + nt * 16 + col;
          v16bf b = bfrag_bf16(Wdhs_bf + (size_t)n * 512 + kb + half * 16);
          acc[nt] = __builtin_amdgcn_wmma_f32_16x16x32_bf16(
              false, a, false, b, (short)0, acc[nt], false, false);
        }
      }
      for (int kb = 0; kb < DECD; kb += 32) {                 // c part, k in [256,512)
        v16bf a = afrag_f32(&c_s[col][0] + kb, half);
#pragma unroll
        for (int nt = 0; nt < 2; ++nt) {
          int n = wid * 32 + nt * 16 + col;
          v16bf b = bfrag_bf16(Wdhs_bf + (size_t)n * 512 + (256 + kb) + half * 16);
          acc[nt] = __builtin_amdgcn_wmma_f32_16x16x32_bf16(
              false, a, false, b, (short)0, acc[nt], false, false);
        }
      }
#pragma unroll
      for (int nt = 0; nt < 2; ++nt) {
        int n = wid * 32 + nt * 16 + col;
        float bias = b_dhs[n];
#pragma unroll
        for (int r = 0; r < 8; ++r) Wd_bf[r + half * 8][n] = f2bf(acc[nt][r] + bias);
      }
    }
    __syncthreads();

    // ---- Phase B: scores[row][t] = sum_e tanh(Wd+Ue)*v  (L2-hot bf16 stream)
    for (int p = tid; p < 16 * TM1; p += 256) {
      int row = p / TM1, t = p - row * TM1;
      const uint4* uep = (const uint4*)(Ue_bf + ((size_t)(r0 + row) * TM1 + t) * ENCD);
      const uint4* wdp = (const uint4*)&Wd_bf[row][0];
      float acc = 0.0f;
#pragma unroll 4
      for (int e8 = 0; e8 < 32; ++e8) {
        uint4 ue = uep[e8];
        uint4 wd = wdp[e8];
        const float* vp = &vv[e8 * 8];
        unsigned int uw[4] = {ue.x, ue.y, ue.z, ue.w};
        unsigned int ww[4] = {wd.x, wd.y, wd.z, wd.w};
#pragma unroll
        for (int q = 0; q < 4; ++q) {
          float2 u2 = bf2x2(uw[q]);
          float2 w2 = bf2x2(ww[q]);
          acc = fmaf(fast_tanh(u2.x + w2.x), vp[2 * q],     acc);
          acc = fmaf(fast_tanh(u2.y + w2.y), vp[2 * q + 1], acc);
        }
      }
      sc[row][t] = acc + bvd;
    }
    __syncthreads();

    // ---- Phase C: softmax over t per row
    if (tid < 16) {
      float mx = -3.4e38f;
      for (int t = 0; t < TM1; ++t) mx = fmaxf(mx, sc[tid][t]);
      float sum = 0.0f;
      for (int t = 0; t < TM1; ++t) {
        float e = fast_exp(sc[tid][t] - mx);
        sc[tid][t] = e;
        sum += e;
      }
      float inv = __builtin_amdgcn_rcpf(sum);
      for (int t = 0; t < TM1; ++t) sc[tid][t] *= inv;
    }
    __syncthreads();

    // ---- Phase D: context[row][e] = sum_t beta * enc
    {
      int row = tid >> 4;
      int e0  = (tid & 15) * 16;
      float a[16];
#pragma unroll
      for (int i = 0; i < 16; ++i) a[i] = 0.0f;
      const unsigned short* ep = enc_bf + (size_t)(r0 + row) * TM1 * ENCD + e0;
      for (int t = 0; t < TM1; ++t) {
        float bta = sc[row][t];
        const uint4* p = (const uint4*)(ep + (size_t)t * ENCD);
        uint4 v0 = p[0], v1 = p[1];
        unsigned int w[8] = {v0.x, v0.y, v0.z, v0.w, v1.x, v1.y, v1.z, v1.w};
#pragma unroll
        for (int q = 0; q < 8; ++q) {
          float2 x = bf2x2(w[q]);
          a[2 * q]     = fmaf(bta, x.x, a[2 * q]);
          a[2 * q + 1] = fmaf(bta, x.y, a[2 * q + 1]);
        }
      }
#pragma unroll
      for (int i = 0; i < 16; ++i) ctx[row][e0 + i] = a[i];
    }
    __syncthreads();

    // ---- Phase E: y_tilde = [context, y] @ W_w^T + b_w
    if (tid < 16) {
      float acc = b_w[0] + y_history[(size_t)(r0 + tid) * TM1 + s] * W_w[ENCD];
      for (int e = 0; e < ENCD; ++e) acc = fmaf(ctx[tid][e], W_w[e], acc);
      ytil[tid] = acc;
    }
    __syncthreads();

    // ---- Phase F: gates = h @ W_hh^T + ...  (8 WMMA tiles/wave: i,f,g,o for
    //      the same 32 columns -> LSTM update fused entirely in registers)
    {
      v8f acc[8];
#pragma unroll
      for (int i = 0; i < 8; ++i) acc[i] = zero8();
      const int j0 = wid * 32;
      for (int kb = 0; kb < DECD; kb += 32) {
        v16bf a = afrag_bf16(&h_bf[col][0] + kb, half);
#pragma unroll
        for (int g = 0; g < 4; ++g) {
#pragma unroll
          for (int nt = 0; nt < 2; ++nt) {
            int n = g * 256 + j0 + nt * 16 + col;
            v16bf b = bfrag_bf16(Whh_bf + (size_t)n * DECD + kb + half * 16);
            acc[g * 2 + nt] = __builtin_amdgcn_wmma_f32_16x16x32_bf16(
                false, a, false, b, (short)0, acc[g * 2 + nt], false, false);
          }
        }
      }
      __syncthreads();   // all waves done reading old h before we overwrite it
#pragma unroll
      for (int nt = 0; nt < 2; ++nt) {
        int j = j0 + nt * 16 + col;
#pragma unroll
        for (int r = 0; r < 8; ++r) {
          int m = r + half * 8;
          float yt = ytil[m];
          float iv = acc[0 + nt][r] + b_ih[j]       + b_hh[j]       + yt * W_ih[j];
          float fv = acc[2 + nt][r] + b_ih[256 + j] + b_hh[256 + j] + yt * W_ih[256 + j];
          float gv = acc[4 + nt][r] + b_ih[512 + j] + b_hh[512 + j] + yt * W_ih[512 + j];
          float ov = acc[6 + nt][r] + b_ih[768 + j] + b_hh[768 + j] + yt * W_ih[768 + j];
          float cn = fast_sigmoid(fv) * c_s[m][j] + fast_sigmoid(iv) * fast_tanh(gv);
          float hn = fast_sigmoid(ov) * fast_tanh(cn);
          c_s[m][j]  = cn;
          h_bf[m][j] = f2bf(hn);
        }
      }
    }
    __syncthreads();
  }

  // ---- Final: y_pred = [h, context] @ W_fc^T + b_fc
  if (tid < 16) {
    float acc = b_fc[0];
    for (int j = 0; j < DECD; ++j) acc = fmaf(bf2f(h_bf[tid][j]), W_fc[j], acc);
    for (int e = 0; e < ENCD; ++e) acc = fmaf(ctx[tid][e], W_fc[DECD + e], acc);
    out[r0 + tid] = acc;
  }
}

// ---------------------------------------------------------------------------
extern "C" void kernel_launch(void* const* d_in, const int* in_sizes, int n_in,
                              void* d_out, int out_size, void* d_ws, size_t ws_size,
                              hipStream_t stream) {
  const float* enc    = (const float*)d_in[0];
  const float* y_hist = (const float*)d_in[1];
  const float* W_vd   = (const float*)d_in[2];
  const float* b_vd   = (const float*)d_in[3];
  const float* W_dhs  = (const float*)d_in[4];
  const float* b_dhs  = (const float*)d_in[5];
  const float* W_Ud   = (const float*)d_in[6];
  const float* b_Ud   = (const float*)d_in[7];
  const float* W_w    = (const float*)d_in[8];
  const float* b_w    = (const float*)d_in[9];
  const float* W_ih   = (const float*)d_in[10];
  const float* W_hh   = (const float*)d_in[11];
  const float* b_ih   = (const float*)d_in[12];
  const float* b_hh   = (const float*)d_in[13];
  const float* W_fc   = (const float*)d_in[14];
  const float* b_fc   = (const float*)d_in[15];
  float* out = (float*)d_out;

  const size_t nEnc = (size_t)BTOT * TM1 * ENCD;     // 33,292,288 elems
  char* ws = (char*)d_ws;
  unsigned short* Ue_bf   = (unsigned short*)ws;  ws += nEnc * 2;
  unsigned short* enc_bf  = (unsigned short*)ws;  ws += nEnc * 2;
  unsigned short* Wdhs_bf = (unsigned short*)ws;  ws += (size_t)ENCD * 2 * DECD * 2;
  unsigned short* Whh_bf  = (unsigned short*)ws;  ws += (size_t)4 * DECD * DECD * 2;
  unsigned short* WUd_bf  = (unsigned short*)ws;  ws += (size_t)ENCD * ENCD * 2;

  f32_to_bf16_kernel<<<2048, 256, 0, stream>>>(enc,   enc_bf,  (int)(nEnc / 8));
  f32_to_bf16_kernel<<<64,   256, 0, stream>>>(W_dhs, Wdhs_bf, ENCD * 2 * DECD / 8);
  f32_to_bf16_kernel<<<128,  256, 0, stream>>>(W_hh,  Whh_bf,  4 * DECD * DECD / 8);
  f32_to_bf16_kernel<<<32,   256, 0, stream>>>(W_Ud,  WUd_bf,  ENCD * ENCD / 8);

  ue_gemm_kernel<<<(BTOT * TM1) / 16, 128, 0, stream>>>(enc_bf, WUd_bf, b_Ud, Ue_bf);

  decoder_kernel<<<BTOT / 16, 256, 0, stream>>>(
      y_hist, Ue_bf, enc_bf, Wdhs_bf, Whh_bf,
      W_vd, b_vd, b_dhs, W_w, b_w, W_ih, b_ih, b_hh, W_fc, b_fc, out);
}
